// Shift_gcn_8495445311793
// MI455X (gfx1250) — matrix operand order repaired and checked by reading, not compiled, not run on hardware
//
#include <hip/hip_runtime.h>
#include <hip/hip_bf16.h>
#include <math.h>

// ---------------------------------------------------------------------------
// Shift-GCN fused kernel for MI455X (gfx1250, wave32, WMMA).
// N=64, C=256, T=64, V=17.  rows = N*T = 4096, feats = V*C = 4352.
// Memory-bound: ~0.86 GB moved -> ~37us floor @ 23.3 TB/s; 9.1 GFLOP of GEMM
// is trivial for v_wmma_f32_16x16x32_f16, so everything is fused into one
// pass over x0 with all intermediates staged in the 320KB WGP LDS.
// ---------------------------------------------------------------------------

typedef __attribute__((ext_vector_type(16))) _Float16 v16h;
typedef __attribute__((ext_vector_type(8)))  float    v8f;
typedef __attribute__((ext_vector_type(4)))  _Float16 v4h;

#define JV   17
#define CC   256
#define TT   64
#define NBAT 64
#define ROWS 16           // t-rows per workgroup (WMMA M)
#define FEAT (JV*CC)      // 4352

// ---- d_ws layout (bytes) ----
#define WS_WH    0                 // 256*256 f16  weight, [c][d]
#define WS_MASK  131072            // 4352 f16     tanh(fm)+1, [w][c]
#define WS_SCALE 139776            // 4352 f32     gamma*rsqrt(var+eps)
#define WS_SHIFT 157184            // 4352 f32     beta - mean*scale

// ---- LDS layout (in _Float16 units) ----
// lx : x0 tile as f16, [c][r*17+v]  (mirrors x0 memory order)   256*272
// yv : GEMM result f16, [(w*16+r)][d] with padded stride 257    17*16*257
//      (stride 257 -> 16*257*2 B = 2056 dwords == 8 mod 64: the
//       (i-d)%17 gather across lanes avoids 17-way bank conflicts)
// awf: per-w A fragments in WMMA 16-bit A 16x32 layout, [kt][lane][16]
#define LX_OFF   0
#define YV_OFF   69632
#define YSTR     257
#define AWF_OFF  (69632 + 69904)   // 139536 halves; byte offset 279072 (16B aligned)
#define LDS_BYTES ((AWF_OFF + 8*32*16) * 2)   // 287,264 B  (< 320 KB)

// ---------------------------------------------------------------------------
__global__ void prep_kernel(const float* __restrict__ lw, const float* __restrict__ fm,
                            const float* __restrict__ g,  const float* __restrict__ be,
                            const float* __restrict__ mean, const float* __restrict__ var,
                            char* __restrict__ ws) {
    int idx = blockIdx.x * blockDim.x + threadIdx.x;
    _Float16* Wh = (_Float16*)(ws + WS_WH);
    _Float16* Mh = (_Float16*)(ws + WS_MASK);
    float*    sc = (float*)(ws + WS_SCALE);
    float*    sh = (float*)(ws + WS_SHIFT);
    if (idx < CC * CC) Wh[idx] = (_Float16)lw[idx];
    if (idx < FEAT) {
        Mh[idx] = (_Float16)(tanhf(fm[idx]) + 1.0f);
        float s = g[idx] * rsqrtf(var[idx] + 1e-5f);
        sc[idx] = s;
        sh[idx] = be[idx] - mean[idx] * s;
    }
}

// ---------------------------------------------------------------------------
__global__ __launch_bounds__(256)
void shiftgcn_kernel(const float* __restrict__ x0, const float* __restrict__ linb,
                     const char* __restrict__ ws, float* __restrict__ out) {
    extern __shared__ _Float16 lds[];
    _Float16* lx  = lds + LX_OFF;
    _Float16* yv  = lds + YV_OFF;
    _Float16* awf = lds + AWF_OFF;

    const _Float16* Wh   = (const _Float16*)(ws + WS_WH);
    const _Float16* Mh   = (const _Float16*)(ws + WS_MASK);
    const float*    bnsc = (const float*)(ws + WS_SCALE);
    const float*    bnsh = (const float*)(ws + WS_SHIFT);

    const int tid  = threadIdx.x;
    const int n    = blockIdx.x >> 2;
    const int t0   = (blockIdx.x & 3) * ROWS;
    const size_t base = (size_t)n * (CC * TT * JV) + (size_t)t0 * JV;  // x0/out tile origin

    // ---------------- Phase 1: x0 tile -> LDS (f32 -> f16) ----------------
    // 256 c-blocks of 272 contiguous floats each; coalesced float4 loads.
    #pragma unroll 4
    for (int k = 0; k < 68; ++k) {
        int idx4 = tid + k * 256;             // 0..17407
        int c    = idx4 / 68;
        int rem4 = idx4 - c * 68;
        const float4 f = *(const float4*)(x0 + base + (size_t)c * (TT * JV) + rem4 * 4);
        v4h h = {(_Float16)f.x, (_Float16)f.y, (_Float16)f.z, (_Float16)f.w};
        *(v4h*)(lx + c * 272 + rem4 * 4) = h;
    }
    __syncthreads();

    const int lane = tid & 31;
    const int wave = tid >> 5;
    const int m    = lane & 15;      // WMMA M row owned by this lane (A) / N col (C,D)
    const int hi   = lane >> 4;      // A K-half selector / C M-half selector
    const float biasA = linb[wave * 32 + m];
    const float biasB = linb[wave * 32 + 16 + m];

    // ---------------- Phase 2: per-w shift-gather + GEMM ----------------
    for (int w = 0; w < JV; ++w) {
        // 2a: build pre-masked A fragments in ISA 16-bit A 16x32 layout:
        //   lanes 0-15: M=lane, K halves {h*8+i} then {16+h*8+i}  (VGPR0..7)
        //   thread tid -> (ktile = tid>>5, lane = tid&31)
        {
            const int k0 = (tid >> 5) * 32;
            v16h a;
            #pragma unroll
            for (int i = 0; i < 16; ++i) {
                int kk = (i < 8) ? (hi * 8 + i) : (16 + hi * 8 + (i - 8));
                int c  = k0 + kk;
                int jv = w + c;  jv -= (jv / JV) * JV;          // (w+c) % 17: shift_in gather
                a[i] = lx[c * 272 + m * JV + jv] * Mh[w * CC + c];
            }
            *(v16h*)(awf + tid * 16) = a;
        }
        __syncthreads();

        // 2b: each wave: M=16 rows x N=32 cols (2 n-tiles), K=256 over 8 k-tiles
        v8f accA = {};
        v8f accB = {};
        const int n0a = wave * 32, n0b = n0a + 16;
        #pragma unroll
        for (int kt = 0; kt < 8; ++kt) {
            const v16h a = *(const v16h*)(awf + (kt * 32 + lane) * 16);   // 2x ds_load_b128
            // B (K=32 x N=16) f16: lane%16 <-> N, elements = K in same table order as A.
            v16h bA, bB;
            #pragma unroll
            for (int i = 0; i < 16; ++i) {
                int kk = (i < 8) ? (hi * 8 + i) : (16 + hi * 8 + (i - 8));
                const int c = kt * 32 + kk;
                bA[i] = Wh[c * CC + n0a + m];
                bB[i] = Wh[c * CC + n0b + m];
            }
            accA = __builtin_amdgcn_wmma_f32_16x16x32_f16(false, a, false, bA,
                                                          (short)0, accA, false, false);
            accB = __builtin_amdgcn_wmma_f32_16x16x32_f16(false, a, false, bB,
                                                          (short)0, accB, false, false);
        }
        // store y[w] (+bias) as f16; C/D layout: lane m = N, VGPR j = row hi*8+j
        #pragma unroll
        for (int j = 0; j < 8; ++j) {
            int r = hi * 8 + j;
            yv[(w * 16 + r) * YSTR + (n0a + m)] = (_Float16)(accA[j] + biasA);
            yv[(w * 16 + r) * YSTR + (n0b + m)] = (_Float16)(accB[j] + biasB);
        }
        __syncthreads();   // protects awf overwrite; final iteration guards phase 3
    }

    // ---------------- Phase 3: output gather + BN + residual + ReLU ------
    // out[n,d,t,v] contiguous along (t,v): fully coalesced float4 stores;
    // x0 residual re-read hits L2 (tile just streamed).
    #pragma unroll 4
    for (int k = 0; k < 68; ++k) {
        int idx4 = tid + k * 256;
        int d    = idx4 / 68;
        int rem4 = (idx4 - d * 68) * 4;
        int dm   = d - (d / JV) * JV;           // d % 17
        const size_t off = base + (size_t)d * (TT * JV) + rem4;
        const float4 xr = *(const float4*)(x0 + off);
        float res[4] = {xr.x, xr.y, xr.z, xr.w};
        float oo[4];
        #pragma unroll
        for (int q = 0; q < 4; ++q) {
            int e = rem4 + q;
            int r = e / JV;
            int i = e - r * JV;
            int wsel = i - dm; if (wsel < 0) wsel += JV;        // shift_out gather
            float z = (float)yv[(wsel * 16 + r) * YSTR + d];
            int f = i * CC + d;
            float val = z * bnsc[f] + bnsh[f] + res[q];
            oo[q] = val > 0.0f ? val : 0.0f;
        }
        float4 o; o.x = oo[0]; o.y = oo[1]; o.z = oo[2]; o.w = oo[3];
        *(float4*)(out + off) = o;
    }
}

// ---------------------------------------------------------------------------
extern "C" void kernel_launch(void* const* d_in, const int* in_sizes, int n_in,
                              void* d_out, int out_size, void* d_ws, size_t ws_size,
                              hipStream_t stream) {
    (void)in_sizes; (void)n_in; (void)out_size; (void)ws_size;
    const float* x0   = (const float*)d_in[0];
    const float* fm   = (const float*)d_in[1];
    const float* lw   = (const float*)d_in[2];
    const float* lb   = (const float*)d_in[3];
    const float* g    = (const float*)d_in[4];
    const float* be   = (const float*)d_in[5];
    const float* mean = (const float*)d_in[6];
    const float* var  = (const float*)d_in[7];
    // d_in[8], d_in[9] (shift index tables) are recomputed analytically.
    char* ws = (char*)d_ws;

    static_assert(LDS_BYTES <= 320 * 1024, "fits CDNA5 WGP LDS");
    // Raise the dynamic-LDS cap to the CDNA5 limit (idempotent, host-side).
    hipFuncSetAttribute(reinterpret_cast<const void*>(&shiftgcn_kernel),
                        hipFuncAttributeMaxDynamicSharedMemorySize, LDS_BYTES);

    prep_kernel<<<256, 256, 0, stream>>>(lw, fm, g, be, mean, var, ws);
    shiftgcn_kernel<<<NBAT * 4, 256, LDS_BYTES, stream>>>(x0, lb, ws, (float*)d_out);
}
